// Pooling_16226386444299
// MI455X (gfx1250) — matrix operand order repaired and checked by reading, not compiled, not run on hardware
//
#include <hip/hip_runtime.h>
#include <hip/hip_bf16.h>

typedef __attribute__((ext_vector_type(16))) _Float16 v16h;
typedef __attribute__((ext_vector_type(8)))  _Float16 v8h;
typedef __attribute__((ext_vector_type(8)))  float    v8f;

#define Nn     50000
#define Ee     800000
#define Cc     6250
#define INTER_ 144
#define SPH_   16
#define INV_   (1.0f/16.0f)

// flat float offsets in d_out (tuple return order)
#define O_NEWX   0
#define O_NEWPOS 800000      // C*128
#define O_NEI    818750      // + C*3
#define O_NEA    2418750     // + 2*E
#define O_NB     4818750     // + 3*E

// ---------------- edge scatter: xsc[c] += x[dst], eac[c] += edge_attr, emit new_edge_index
__global__ __launch_bounds__(256)
void k_edge(const float* __restrict__ x, const int* __restrict__ ei,
            const float* __restrict__ ea, float* __restrict__ xsc,
            float* __restrict__ eac, float* __restrict__ out) {
  const int wid  = threadIdx.x >> 5;
  const int lane = threadIdx.x & 31;
  const int e = blockIdx.x * 8 + wid;
  if (e >= Ee) return;
  const int s  = ei[e];
  const int d  = ei[Ee + e];
  const int cs = s >> 3;
  const int cd = d >> 3;
  const float4 v = ((const float4*)(x + (size_t)d * 128))[lane];
  float* p = xsc + (size_t)cs * 128 + lane * 4;
  atomicAdd(p + 0, v.x);
  atomicAdd(p + 1, v.y);
  atomicAdd(p + 2, v.z);
  atomicAdd(p + 3, v.w);
  if (lane < 3) atomicAdd(eac + (size_t)cs * 3 + lane, ea[(size_t)e * 3 + lane]);
  if (lane == 0) {
    out[O_NEI + e]      = (float)cs;
    out[O_NEI + Ee + e] = (float)cd;
  }
}

// ---------------- group-of-8 pooling of x -> xc (C x 128)
__global__ __launch_bounds__(256)
void k_pool_x(const float* __restrict__ x, float* __restrict__ xc) {
  const int idx = blockIdx.x * 256 + threadIdx.x;
  if (idx >= Cc * 128) return;
  const int c = idx >> 7;
  const int f = idx & 127;
  const float* p = x + (size_t)c * 8 * 128 + f;
  float s = 0.0f;
#pragma unroll
  for (int j = 0; j < 8; ++j) s += p[(size_t)j * 128];
  xc[idx] = s;
}

// ---------------- new_pos, gsum, new_batch
__global__ __launch_bounds__(256)
void k_posg(const float* __restrict__ pos, const int* __restrict__ batch,
            float* __restrict__ gsum, float* __restrict__ out) {
  const int c = blockIdx.x * 256 + threadIdx.x;
  if (c >= Cc) return;
  float s0 = 0, s1 = 0, s2 = 0;
#pragma unroll
  for (int j = 0; j < 8; ++j) {
    const float* p = pos + (size_t)(8 * c + j) * 3;
    s0 += p[0]; s1 += p[1]; s2 += p[2];
  }
  const float n0 = s0 * 0.125f, n1 = s1 * 0.125f, n2 = s2 * 0.125f;
  out[O_NEWPOS + (size_t)c * 3 + 0] = n0;
  out[O_NEWPOS + (size_t)c * 3 + 1] = n1;
  out[O_NEWPOS + (size_t)c * 3 + 2] = n2;
  float g0 = 0, g1 = 0, g2 = 0;
#pragma unroll
  for (int j = 0; j < 8; ++j) {
    const float* p = pos + (size_t)(8 * c + j) * 3;
    g0 += p[0] - n0; g1 += p[1] - n1; g2 += p[2] - n2;
  }
  gsum[(size_t)c * 3 + 0] = g0;
  gsum[(size_t)c * 3 + 1] = g1;
  gsum[(size_t)c * 3 + 2] = g2;
  int mb = batch[8 * c];
#pragma unroll
  for (int j = 1; j < 8; ++j) { int b = batch[8 * c + j]; mb = (b > mb) ? b : mb; }
  out[O_NB + c] = (float)mb;
}

// helper: pack two contiguous 8-float runs into a 16-half A/B-style fragment
__device__ __forceinline__ v16h pack_two_runs(const float* r0, const float* r1, float scale) {
  float4 u0 = *(const float4*)(r0);
  float4 u1 = *(const float4*)(r0 + 4);
  float4 u2 = *(const float4*)(r1);
  float4 u3 = *(const float4*)(r1 + 4);
  v16h a;
  a[0]  = (_Float16)(u0.x * scale); a[1]  = (_Float16)(u0.y * scale);
  a[2]  = (_Float16)(u0.z * scale); a[3]  = (_Float16)(u0.w * scale);
  a[4]  = (_Float16)(u1.x * scale); a[5]  = (_Float16)(u1.y * scale);
  a[6]  = (_Float16)(u1.z * scale); a[7]  = (_Float16)(u1.w * scale);
  a[8]  = (_Float16)(u2.x * scale); a[9]  = (_Float16)(u2.y * scale);
  a[10] = (_Float16)(u2.z * scale); a[11] = (_Float16)(u2.w * scale);
  a[12] = (_Float16)(u3.x * scale); a[13] = (_Float16)(u3.y * scale);
  a[14] = (_Float16)(u3.z * scale); a[15] = (_Float16)(u3.w * scale);
  return a;
}

// ---------------- fused WMMA GEMM: featsum (LDS) then new_x
__global__ __launch_bounds__(256)
void k_gemm(const float* __restrict__ xc, const float* __restrict__ xsc,
            const float* __restrict__ eac, const float* __restrict__ gsum,
            const float* __restrict__ W0, const float* __restrict__ W1,
            const float* __restrict__ We, const float* __restrict__ Wg,
            const float* __restrict__ Wge, float* __restrict__ out) {
  __shared__ _Float16 lfeat[16 * 128];
  const int lane = threadIdx.x & 31;
  const int wave = threadIdx.x >> 5;
  const int g    = lane >> 4;   // lane half-group
  const int lm   = lane & 15;   // M for A, N for B/D
  const int m0   = blockIdx.x * 16;
  const int n0   = wave * 16;
  const int colF = n0 + lm;

  // Clamp instead of predicate: OOB rows only pollute their own (guarded) outputs.
  const int rowA = m0 + lm;
  const int rA   = (rowA < Cc) ? rowA : (Cc - 1);

  // Phase 1: featsum = xc@W0[:,16:] + inv*xsc@W1[:,16:]  (+ inv*eac@We[16:] epilogue)
  v8f acc = {};
#pragma unroll
  for (int k0 = 0; k0 < 128; k0 += 32) {
    const float* pA0 = xc  + (size_t)rA * 128 + k0 + 8 * g;
    const float* pA1 = xsc + (size_t)rA * 128 + k0 + 8 * g;
    v16h a0 = pack_two_runs(pA0, pA0 + 16, 1.0f);
    v16h a1 = pack_two_runs(pA1, pA1 + 16, INV_);
    v16h b0, b1;
#pragma unroll
    for (int e2 = 0; e2 < 16; ++e2) {
      const int K = k0 + 8 * g + (e2 & 7) + ((e2 >> 3) << 4);
      b0[e2] = (_Float16)W0[(size_t)K * INTER_ + SPH_ + colF];
      b1[e2] = (_Float16)W1[(size_t)K * INTER_ + SPH_ + colF];
    }
    acc = __builtin_amdgcn_wmma_f32_16x16x32_f16(false, a0, false, b0, (short)0, acc, false, false);
    acc = __builtin_amdgcn_wmma_f32_16x16x32_f16(false, a1, false, b1, (short)0, acc, false, false);
  }

#pragma unroll
  for (int r = 0; r < 8; ++r) {
    const int M   = r + 8 * g;
    const int row = m0 + M;
    const int rc  = (row < Cc) ? row : (Cc - 1);
    float fs = acc[r]
             + INV_ * (eac[(size_t)rc * 3 + 0] * We[0 * INTER_ + SPH_ + colF]
                     + eac[(size_t)rc * 3 + 1] * We[1 * INTER_ + SPH_ + colF]
                     + eac[(size_t)rc * 3 + 2] * We[2 * INTER_ + SPH_ + colF]);
    lfeat[M * 128 + colF] = (_Float16)fs;
  }
  __syncthreads();

  // Phase 2: new_x = inv*(featsum@Wg + gsum@Wge)
  v8f acc2 = {};
#pragma unroll
  for (int k0 = 0; k0 < 128; k0 += 32) {
    const _Float16* pL = lfeat + lm * 128 + k0 + 8 * g;
    v8h r1 = *(const v8h*)(pL);
    v8h r2 = *(const v8h*)(pL + 16);
    v16h a;
#pragma unroll
    for (int e2 = 0; e2 < 8; ++e2) { a[e2] = r1[e2]; a[8 + e2] = r2[e2]; }
    v16h b;
#pragma unroll
    for (int e2 = 0; e2 < 16; ++e2) {
      const int K = k0 + 8 * g + (e2 & 7) + ((e2 >> 3) << 4);
      b[e2] = (_Float16)Wg[(size_t)K * 128 + colF];
    }
    acc2 = __builtin_amdgcn_wmma_f32_16x16x32_f16(false, a, false, b, (short)0, acc2, false, false);
  }

#pragma unroll
  for (int r = 0; r < 8; ++r) {
    const int M   = r + 8 * g;
    const int row = m0 + M;
    const int rc  = (row < Cc) ? row : (Cc - 1);
    const float val = acc2[r]
                    + gsum[(size_t)rc * 3 + 0] * Wge[0 * 128 + colF]
                    + gsum[(size_t)rc * 3 + 1] * Wge[1 * 128 + colF]
                    + gsum[(size_t)rc * 3 + 2] * Wge[2 * 128 + colF];
    if (row < Cc) out[O_NEWX + (size_t)row * 128 + colF] = val * INV_;
  }
}

// ---------------- new_edge_attr = new_pos[cd] - new_pos[cs]
__global__ __launch_bounds__(256)
void k_edge_attr(const int* __restrict__ ei, float* __restrict__ out) {
  const int e = blockIdx.x * 256 + threadIdx.x;
  if (e >= Ee) return;
  const int cs = ei[e] >> 3;
  const int cd = ei[Ee + e] >> 3;
  const float* np = out + O_NEWPOS;
#pragma unroll
  for (int k = 0; k < 3; ++k)
    out[O_NEA + (size_t)e * 3 + k] = np[(size_t)cd * 3 + k] - np[(size_t)cs * 3 + k];
}

extern "C" void kernel_launch(void* const* d_in, const int* in_sizes, int n_in,
                              void* d_out, int out_size, void* d_ws, size_t ws_size,
                              hipStream_t stream) {
  (void)in_sizes; (void)n_in; (void)out_size; (void)ws_size;
  const float* x     = (const float*)d_in[0];
  const float* pos   = (const float*)d_in[1];
  const int*   ei    = (const int*)d_in[2];
  const float* ea    = (const float*)d_in[3];
  const int*   batch = (const int*)d_in[4];
  const float* W0    = (const float*)d_in[5];
  const float* W1    = (const float*)d_in[6];
  const float* We    = (const float*)d_in[7];
  const float* Wg    = (const float*)d_in[8];
  const float* Wge   = (const float*)d_in[9];
  float* out = (float*)d_out;

  float* xsc  = (float*)d_ws;                       // C*128   (16B aligned)
  float* eac  = xsc  + (size_t)Cc * 128;            // C*3
  float* gsum = eac  + (size_t)Cc * 3;              // C*3
  float* xc   = gsum + (size_t)Cc * 3;              // C*128   (byte off 3,350,000 -> 16B aligned)

  // zero the atomic accumulators (xsc + eac are contiguous)
  hipMemsetAsync(d_ws, 0, ((size_t)Cc * 128 + (size_t)Cc * 3) * sizeof(float), stream);

  k_edge     <<<Ee / 8,                 256, 0, stream>>>(x, ei, ea, xsc, eac, out);
  k_pool_x   <<<(Cc * 128 + 255) / 256, 256, 0, stream>>>(x, xc);
  k_posg     <<<(Cc + 255) / 256,       256, 0, stream>>>(pos, batch, gsum, out);
  k_gemm     <<<(Cc + 15) / 16,         256, 0, stream>>>(xc, xsc, eac, gsum,
                                                          W0, W1, We, Wg, Wge, out);
  k_edge_attr<<<(Ee + 255) / 256,       256, 0, stream>>>(ei, out);
}
